// PLETradingModel_5952824672297
// MI455X (gfx1250) — matrix operand order repaired and checked by reading, not compiled, not run on hardware
//
#include <hip/hip_runtime.h>
#include <hip/hip_bf16.h>
#include <stdint.h>

// PLE trading model, gfx1250: TDM (tensor_load_to_lds) feeds LDS, f16 WMMA consumes.
// Workspace (~131 MB): h16, out_all, wgate, x16, and pre-transposed f16 weights.

typedef __attribute__((ext_vector_type(16))) _Float16 v16h;
typedef __attribute__((ext_vector_type(8)))  _Float16 v8h;
typedef __attribute__((ext_vector_type(8)))  float    v8f;
typedef __attribute__((ext_vector_type(4)))  unsigned int u32x4;
typedef __attribute__((ext_vector_type(8)))  int i32x8;
typedef __attribute__((ext_vector_type(4)))  int i32x4;

#define DEVI __device__ __forceinline__

#if defined(__has_builtin)
#  if __has_builtin(__builtin_amdgcn_tensor_load_to_lds)
#    define USE_TDM 1
#  endif
#endif
#ifndef USE_TDM
#  define USE_TDM 0
#endif

namespace {
constexpr int Bsz = 32768;
constexpr int F   = 416;
constexpr int D   = 128;
constexpr int Hd  = 256;
constexpr int E   = 11;
constexpr int T   = 4;
constexpr int G   = 5;
constexpr int LVL = 2;
constexpr int THd = 64;
constexpr int O   = 100;
constexpr int OP  = 112;          // O padded to 7 WMMA tiles
constexpr int MB  = 128;          // batch rows per workgroup
constexpr float EPS = 1e-5f;
}

DEVI v8f wmma_f16(v16h a, v16h b, v8f c) {
  return __builtin_amdgcn_wmma_f32_16x16x32_f16(false, a, false, b, (short)0, c,
                                                false, false);
}

DEVI v16h cat16(v8h lo, v8h hi) {
  return __builtin_shufflevector(lo, hi, 0, 1, 2, 3, 4, 5, 6, 7,
                                 8, 9, 10, 11, 12, 13, 14, 15);
}

// A-frag (16x32 f16): lane m = lane&15; elems 0..7 = K koff..+7, 8..15 = K koff+16..+23,
// koff = (lane>=16)?8:0.  sA row-major [row][k], stride in halfs (16B-aligned rows).
DEVI v16h fragA(const _Float16* sA, int strideA, int m0, int kb, int lane) {
  const _Float16* p = sA + (m0 + (lane & 15)) * strideA + kb + ((lane & 16) ? 8 : 0);
  v8h lo = *(const v8h*)(p);
  v8h hi = *(const v8h*)(p + 16);
  return cat16(lo, hi);
}

// B-frag (32x16 f16): lane n = lane&15; elems 0..7 = K koff..+7, 8..15 = K koff+16..+23,
// koff=(lane>=16)?8:0.  sB transposed [n][k].
DEVI v16h fragB(const _Float16* sB, int strideB, int n0, int lane) {
  const _Float16* p = sB + (n0 + (lane & 15)) * strideB + ((lane & 16) ? 8 : 0);
  v8h lo = *(const v8h*)(p);
  v8h hi = *(const v8h*)(p + 16);
  return cat16(lo, hi);
}

DEVI v8f zero8() {
  v8f z = {0.f, 0.f, 0.f, 0.f, 0.f, 0.f, 0.f, 0.f};
  return z;
}

DEVI float gelu_exact(float x) {
  return 0.5f * x * (1.0f + erff(x * 0.70710678118654752f));
}

DEVI float half16_sum(float v) {
  v += __shfl_xor(v, 1, 32);
  v += __shfl_xor(v, 2, 32);
  v += __shfl_xor(v, 4, 32);
  v += __shfl_xor(v, 8, 32);
  return v;
}

// Low 32 bits of a generic pointer to LDS = LDS byte offset (ISA 10.2 aperture rule).
DEVI unsigned lds_off(const void* p) {
  return (unsigned)(unsigned long long)(uintptr_t)p;
}

// 2D TDM: load nRows x kElems f16 tile (row stride strideElems in memory) into LDS
// at ldsDst with row stride ldsStrideH halfs, using TDM pad_interval/pad_amount.
// Issue from ONE wave only (EXEC is ignored by TDM; one DMA per workgroup).
DEVI void tdm_load_2d(void* ldsDst, const _Float16* gsrc, int kElems, int nRows,
                      int strideElems, int ldsStrideH) {
#if USE_TDM
  int padInt = 31 - __builtin_clz((unsigned)(kElems * 2 / 8));  // row bytes / 8 -> log2
  int padAmt = (ldsStrideH - kElems) / 2 - 1;                   // pad dwords - 1
  unsigned long long ga = (unsigned long long)(uintptr_t)gsrc;
  u32x4 g0;
  g0[0] = 1u;                                           // count=1, user mode
  g0[1] = lds_off(ldsDst);                              // lds_addr (bytes)
  g0[2] = (unsigned)ga;                                 // global_addr[31:0]
  g0[3] = (unsigned)((ga >> 32) & 0x1ffffffull) | (2u << 30);  // addr[56:32], type=2
  i32x8 g1;
  g1[0] = (1 << 16) | (1 << 20) | (padInt << 22) | (padAmt << 25);  // 2B elems, pad
  g1[1] = (kElems & 0xffff) << 16;                      // tensor_dim0[15:0]
  g1[2] = (((unsigned)kElems >> 16) & 0xffff) | ((nRows & 0xffff) << 16);
  g1[3] = (((unsigned)nRows >> 16) & 0xffff) | ((kElems & 0xffff) << 16); // tile_dim0
  g1[4] = (nRows & 0xffff);                             // tile_dim1 (tile_dim2=0)
  g1[5] = strideElems;                                  // tensor_dim0_stride[31:0]
  g1[6] = 0;
  g1[7] = 0;
  i32x4 z4 = {0, 0, 0, 0};
  i32x8 z8 = {0, 0, 0, 0, 0, 0, 0, 0};
  // clang-23 / therock-10.0 lane: 6-arg form (g0, g1, g2, g3, extra, cpol)
  __builtin_amdgcn_tensor_load_to_lds(g0, g1, z4, z4, z8, 0);
#else
  _Float16* d = (_Float16*)ldsDst;
  int lane = threadIdx.x & 31;
  for (int i = lane; i < nRows * kElems; i += 32) {
    int r = i / kElems, c = i % kElems;
    d[r * ldsStrideH + c] = gsrc[(size_t)r * strideElems + c];
  }
#endif
}

DEVI void tdm_wait(int n) {
#if USE_TDM
  if (n == 0) __builtin_amdgcn_s_wait_tensorcnt(0);
  else if (n == 1) __builtin_amdgcn_s_wait_tensorcnt(1);
  else __builtin_amdgcn_s_wait_tensorcnt(2);
#else
  (void)n;
#endif
}

// ---------------------------------------------------------------------------
// One-time conversion kernels: f32 -> f16, with transpose for B-operands.
// ---------------------------------------------------------------------------
__global__ void k_cvt_copy(const float* __restrict__ src, _Float16* __restrict__ dst,
                           int n) {
  int i = blockIdx.x * blockDim.x + threadIdx.x;
  if (i < n) dst[i] = (_Float16)src[i];
}

// src [m][rows(k)][cols(n)] -> dst [m][cols(n)][rows(k)]
__global__ void k_cvt_tr(const float* __restrict__ src, _Float16* __restrict__ dst,
                         int mats, int rows, int cols) {
  int i = blockIdx.x * blockDim.x + threadIdx.x;
  if (i < mats * rows * cols) {
    int m = i / (rows * cols), r = i % (rows * cols);
    int n = r / rows, k = r % rows;
    dst[i] = (_Float16)src[((size_t)m * rows + k) * cols + n];
  }
}

// Wt2 (T,THd,O) -> [T][OP][THd] transposed, zero-padded n in [O, OP)
__global__ void k_cvt_wt2(const float* __restrict__ src, _Float16* __restrict__ dst) {
  int i = blockIdx.x * blockDim.x + threadIdx.x;
  if (i < T * OP * THd) {
    int t = i / (OP * THd), r = i % (OP * THd);
    int n = r / THd, k = r % THd;
    dst[i] = (_Float16)((n < O) ? src[((size_t)t * THd + k) * O + n] : 0.0f);
  }
}

// ---------------------------------------------------------------------------
// Kernel 1: h = layernorm(gelu(x @ Wp + bp), gp, bep) -> h16 (B x D, f16)
// ---------------------------------------------------------------------------
__global__ __launch_bounds__(256) void k_proj(
    const _Float16* __restrict__ x16, const _Float16* __restrict__ Wpt,
    const float* __restrict__ bp, const float* __restrict__ gp,
    const float* __restrict__ bep, _Float16* __restrict__ h16) {
  __shared__ __align__(16) _Float16 sA[2][MB * 40];
  __shared__ __align__(16) _Float16 sB[2][D * 40];

  const int tid = threadIdx.x;
  const int lane = tid & 31;
  const int wv = tid >> 5;
  const int m0 = wv * 16;
  const int rowBase = blockIdx.x * MB;
  const int nSteps = F / 32;  // 13

  v8f acc[8];
#pragma unroll
  for (int i = 0; i < 8; ++i) acc[i] = zero8();

  if (wv == 0) {
    tdm_load_2d(sA[0], x16 + (size_t)rowBase * F, 32, MB, F, 40);
    tdm_load_2d(sB[0], Wpt, 32, D, F, 40);
  }
  for (int ks = 0; ks < nSteps; ++ks) {
    if (wv == 0) {
      if (ks + 1 < nSteps) {
        int kb = (ks + 1) * 32;
        tdm_load_2d(sA[(ks + 1) & 1], x16 + (size_t)rowBase * F + kb, 32, MB, F, 40);
        tdm_load_2d(sB[(ks + 1) & 1], Wpt + kb, 32, D, F, 40);
        tdm_wait(2);
      } else {
        tdm_wait(0);
      }
    }
    __syncthreads();
    v16h a = fragA(sA[ks & 1], 40, m0, 0, lane);
#pragma unroll
    for (int nt = 0; nt < 8; ++nt)
      acc[nt] = wmma_f16(a, fragB(sB[ks & 1], 40, nt * 16, lane), acc[nt]);
    __syncthreads();
  }

#pragma unroll
  for (int v = 0; v < 8; ++v) {
    int rl = m0 + v + ((lane & 16) ? 8 : 0);
    float gv[8];
    float s = 0.f, sq = 0.f;
#pragma unroll
    for (int nt = 0; nt < 8; ++nt) {
      int nn = nt * 16 + (lane & 15);
      float g = gelu_exact(acc[nt][v] + bp[nn]);
      gv[nt] = g;
      s += g;
      sq += g * g;
    }
    s = half16_sum(s);
    sq = half16_sum(sq);
    float mu = s * (1.f / 128.f);
    float rs = rsqrtf(sq * (1.f / 128.f) - mu * mu + EPS);
#pragma unroll
    for (int nt = 0; nt < 8; ++nt) {
      int nn = nt * 16 + (lane & 15);
      float o = (gv[nt] - mu) * rs * gp[nn] + bep[nn];
      h16[(size_t)(rowBase + rl) * D + nn] = (_Float16)o;
    }
  }
}

// ---------------------------------------------------------------------------
// Kernel 2/3: one PLE level. Weights pre-offset to level l; W1t/W2t are f16
// transposed [e][n][k].  FINAL==0: h <- sum_e coef_e * LN(expert_e(h)).
// FINAL==1: out_all[b][e][:] = LN(expert_e(h)), wgate = per-task softmax.
// ---------------------------------------------------------------------------
template <int FINAL>
__global__ __launch_bounds__(256) void k_ple(
    const _Float16* __restrict__ W1t, const float* __restrict__ b1,
    const _Float16* __restrict__ W2t, const float* __restrict__ b2,
    const float* __restrict__ g_ln, const float* __restrict__ be_ln,
    const float* __restrict__ Wg, const float* __restrict__ bg,
    _Float16* __restrict__ h16, _Float16* __restrict__ out_all,
    float* __restrict__ wgate) {
  __shared__ __align__(16) _Float16 sA[MB * 136];    // h tile 128x128
  __shared__ __align__(16) _Float16 sZ[MB * 136];    // gelu(z) half-tile
  __shared__ __align__(16) _Float16 sBd[2][D * 40];  // weight chunks (ping-pong)
  __shared__ float sCoef[MB * 12];

  const int tid = threadIdx.x;
  const int lane = tid & 31;
  const int wv = tid >> 5;
  const int m0 = wv * 16;
  const int rowBase = blockIdx.x * MB;

  if (wv == 0) {
    tdm_load_2d(sA, h16 + (size_t)rowBase * D, 128, 128, D, 136);
    tdm_wait(0);
  }
  __syncthreads();

  // gating: one thread per row
  if (tid < MB) {
    float lg[T * G];
#pragma unroll
    for (int i = 0; i < T * G; ++i) lg[i] = bg[i];
    for (int d = 0; d < D; ++d) {
      float hv = (float)sA[tid * 136 + d];
#pragma unroll
      for (int t = 0; t < T; ++t)
#pragma unroll
        for (int g = 0; g < G; ++g)
          lg[t * G + g] += hv * Wg[((size_t)t * D + d) * G + g];
    }
    float cf[E];
#pragma unroll
    for (int e = 0; e < E; ++e) cf[e] = 0.f;
#pragma unroll
    for (int t = 0; t < T; ++t) {
      float mx = lg[t * G];
#pragma unroll
      for (int g = 1; g < G; ++g) mx = fmaxf(mx, lg[t * G + g]);
      float w[G];
      float s = 0.f;
#pragma unroll
      for (int g = 0; g < G; ++g) {
        w[g] = expf(lg[t * G + g] - mx);
        s += w[g];
      }
      float inv = 1.0f / s;
#pragma unroll
      for (int g = 0; g < G; ++g) {
        float wn = w[g] * inv;
        int e = (g < 3) ? g : (2 * t + g);  // IDX[t][g] = {0,1,2,3+2t,4+2t}
        if (FINAL) {
          wgate[(size_t)(rowBase + tid) * (T * G) + t * G + g] = wn;
        } else {
          cf[e] += wn;
        }
      }
    }
    if (!FINAL) {
#pragma unroll
      for (int e = 0; e < E; ++e) sCoef[tid * 12 + e] = cf[e] * 0.25f;
    }
  }
  __syncthreads();

  v8f accH[8];
#pragma unroll
  for (int i = 0; i < 8; ++i) accH[i] = zero8();

  for (int e = 0; e < E; ++e) {
    const _Float16* W1e = W1t + (size_t)e * Hd * D;  // [n(256)][k(128)]
    const _Float16* W2e = W2t + (size_t)e * D * Hd;  // [n(128)][k(256)]

    v8f accO[8];
#pragma unroll
    for (int i = 0; i < 8; ++i) accO[i] = zero8();

    for (int hh = 0; hh < 2; ++hh) {
      const int nbase = hh * 128;
      v8f accZ[8];
#pragma unroll
      for (int i = 0; i < 8; ++i) accZ[i] = zero8();

      // GEMM1: z_half = h(128x128) @ W1e rows [nbase, nbase+128)
      if (wv == 0) tdm_load_2d(sBd[0], W1e + (size_t)nbase * D, 32, 128, D, 40);
      for (int ks = 0; ks < 4; ++ks) {
        if (wv == 0) {
          if (ks < 3) {
            tdm_load_2d(sBd[(ks + 1) & 1], W1e + (size_t)nbase * D + (ks + 1) * 32,
                        32, 128, D, 40);
            tdm_wait(1);
          } else {
            tdm_wait(0);
          }
        }
        __syncthreads();
        v16h a = fragA(sA, 136, m0, ks * 32, lane);
#pragma unroll
        for (int nt = 0; nt < 8; ++nt)
          accZ[nt] = wmma_f16(a, fragB(sBd[ks & 1], 40, nt * 16, lane), accZ[nt]);
        __syncthreads();
      }

      // gelu(z + b1) -> sZ
#pragma unroll
      for (int nt = 0; nt < 8; ++nt) {
        int nn = nt * 16 + (lane & 15);
        float bb = b1[e * Hd + nbase + nn];
#pragma unroll
        for (int v = 0; v < 8; ++v) {
          int rl = m0 + v + ((lane & 16) ? 8 : 0);
          sZ[rl * 136 + nn] = (_Float16)gelu_exact(accZ[nt][v] + bb);
        }
      }
      __syncthreads();

      // GEMM2 partial: accO += z_half @ W2e[:, nbase:nbase+128]
      if (wv == 0) tdm_load_2d(sBd[0], W2e + nbase, 32, 128, Hd, 40);
      for (int ks = 0; ks < 4; ++ks) {
        if (wv == 0) {
          if (ks < 3) {
            tdm_load_2d(sBd[(ks + 1) & 1], W2e + nbase + (ks + 1) * 32, 32, 128, Hd, 40);
            tdm_wait(1);
          } else {
            tdm_wait(0);
          }
        }
        __syncthreads();
        v16h a = fragA(sZ, 136, m0, ks * 32, lane);
#pragma unroll
        for (int nt = 0; nt < 8; ++nt)
          accO[nt] = wmma_f16(a, fragB(sBd[ks & 1], 40, nt * 16, lane), accO[nt]);
        __syncthreads();
      }
    }

    // epilogue: +b2 +residual, layernorm, combine/store
#pragma unroll
    for (int v = 0; v < 8; ++v) {
      int rl = m0 + v + ((lane & 16) ? 8 : 0);
      float vals[8];
      float s = 0.f, sq = 0.f;
#pragma unroll
      for (int nt = 0; nt < 8; ++nt) {
        int nn = nt * 16 + (lane & 15);
        float z2 = accO[nt][v] + b2[e * D + nn] + (float)sA[rl * 136 + nn];
        vals[nt] = z2;
        s += z2;
        sq += z2 * z2;
      }
      s = half16_sum(s);
      sq = half16_sum(sq);
      float mu = s * (1.f / 128.f);
      float rs = rsqrtf(sq * (1.f / 128.f) - mu * mu + EPS);
      if (FINAL) {
#pragma unroll
        for (int nt = 0; nt < 8; ++nt) {
          int nn = nt * 16 + (lane & 15);
          float o = (vals[nt] - mu) * rs * g_ln[e * D + nn] + be_ln[e * D + nn];
          out_all[((size_t)(rowBase + rl) * E + e) * D + nn] = (_Float16)o;
        }
      } else {
        float c = sCoef[rl * 12 + e];
#pragma unroll
        for (int nt = 0; nt < 8; ++nt) {
          int nn = nt * 16 + (lane & 15);
          float o = (vals[nt] - mu) * rs * g_ln[e * D + nn] + be_ln[e * D + nn];
          accH[nt][v] += c * o;
        }
      }
    }
  }

  if (!FINAL) {
    __syncthreads();
#pragma unroll
    for (int nt = 0; nt < 8; ++nt)
#pragma unroll
      for (int v = 0; v < 8; ++v) {
        int rl = m0 + v + ((lane & 16) ? 8 : 0);
        int nn = nt * 16 + (lane & 15);
        h16[(size_t)(rowBase + rl) * D + nn] = (_Float16)accH[nt][v];
      }
  }
}

// ---------------------------------------------------------------------------
// Kernel 4: per-task gated combine + two-layer head -> out (4, B, 100) f32.
// ---------------------------------------------------------------------------
__global__ __launch_bounds__(256) void k_heads(
    const _Float16* __restrict__ out_all, const float* __restrict__ wgate,
    const _Float16* __restrict__ Wt1t, const float* __restrict__ bt1,
    const _Float16* __restrict__ Wt2t, const float* __restrict__ bt2,
    float* __restrict__ outp) {
  __shared__ __align__(16) _Float16 sG[MB * 136];
  __shared__ __align__(16) _Float16 sB[2][OP * 40];
  __shared__ __align__(16) _Float16 sH2[MB * 72];

  const int tid = threadIdx.x;
  const int lane = tid & 31;
  const int wv = tid >> 5;
  const int m0 = wv * 16;
  const int rowBase = blockIdx.x * MB;

  for (int t = 0; t < T; ++t) {
    __syncthreads();
    for (int i = tid; i < MB * D; i += 256) {
      int r = i >> 7, d = i & 127;
      size_t rb = (size_t)(rowBase + r);
      float acc = 0.f;
#pragma unroll
      for (int g = 0; g < G; ++g) {
        int e = (g < 3) ? g : (2 * t + g);
        acc += wgate[rb * (T * G) + t * G + g] * (float)out_all[(rb * E + e) * D + d];
      }
      sG[r * 136 + d] = (_Float16)acc;
    }
    __syncthreads();

    // GEMM1: hid = gelu(gated(128x128) @ Wt1[t](128x64) + bt1)
    const _Float16* W1p = Wt1t + (size_t)t * THd * D;  // [n(64)][k(128)]
    v8f acc1[4];
#pragma unroll
    for (int i = 0; i < 4; ++i) acc1[i] = zero8();
    if (wv == 0) tdm_load_2d(sB[0], W1p, 32, THd, D, 40);
    for (int ks = 0; ks < 4; ++ks) {
      if (wv == 0) {
        if (ks < 3) {
          tdm_load_2d(sB[(ks + 1) & 1], W1p + (ks + 1) * 32, 32, THd, D, 40);
          tdm_wait(1);
        } else {
          tdm_wait(0);
        }
      }
      __syncthreads();
      v16h a = fragA(sG, 136, m0, ks * 32, lane);
#pragma unroll
      for (int nt = 0; nt < 4; ++nt)
        acc1[nt] = wmma_f16(a, fragB(sB[ks & 1], 40, nt * 16, lane), acc1[nt]);
      __syncthreads();
    }
#pragma unroll
    for (int nt = 0; nt < 4; ++nt) {
      int nn = nt * 16 + (lane & 15);
      float bb = bt1[t * THd + nn];
#pragma unroll
      for (int v = 0; v < 8; ++v) {
        int rl = m0 + v + ((lane & 16) ? 8 : 0);
        sH2[rl * 72 + nn] = (_Float16)gelu_exact(acc1[nt][v] + bb);
      }
    }
    __syncthreads();

    // GEMM2: out = hid(128x64) @ Wt2[t](64x100 padded 112) + bt2
    const _Float16* W2p = Wt2t + (size_t)t * OP * THd;  // [n(112)][k(64)]
    v8f acc2[7];
#pragma unroll
    for (int i = 0; i < 7; ++i) acc2[i] = zero8();
    if (wv == 0) tdm_load_2d(sB[0], W2p, 32, OP, THd, 40);
    for (int ks = 0; ks < 2; ++ks) {
      if (wv == 0) {
        if (ks < 1) {
          tdm_load_2d(sB[1], W2p + 32, 32, OP, THd, 40);
          tdm_wait(1);
        } else {
          tdm_wait(0);
        }
      }
      __syncthreads();
      v16h a = fragA(sH2, 72, m0, ks * 32, lane);
#pragma unroll
      for (int nt = 0; nt < 7; ++nt)
        acc2[nt] = wmma_f16(a, fragB(sB[ks & 1], 40, nt * 16, lane), acc2[nt]);
      __syncthreads();
    }

#pragma unroll
    for (int nt = 0; nt < 7; ++nt) {
      int nn = nt * 16 + (lane & 15);
      if (nn < O) {
        float bb = bt2[t * O + nn];
#pragma unroll
        for (int v = 0; v < 8; ++v) {
          int rl = m0 + v + ((lane & 16) ? 8 : 0);
          float o = acc2[nt][v] + bb;
          if (t == 0) o = 1.0f / (1.0f + expf(-o));
          outp[((size_t)t * Bsz + rowBase + rl) * O + nn] = o;
        }
      }
    }
  }
}

// ---------------------------------------------------------------------------
extern "C" void kernel_launch(void* const* d_in, const int* in_sizes, int n_in,
                              void* d_out, int out_size, void* d_ws,
                              size_t ws_size, hipStream_t stream) {
  (void)in_sizes; (void)n_in; (void)out_size; (void)ws_size;
  const float* x    = (const float*)d_in[0];
  const float* Wp   = (const float*)d_in[1];
  const float* bp   = (const float*)d_in[2];
  const float* gp   = (const float*)d_in[3];
  const float* bep  = (const float*)d_in[4];
  const float* W1   = (const float*)d_in[5];
  const float* b1   = (const float*)d_in[6];
  const float* W2   = (const float*)d_in[7];
  const float* b2   = (const float*)d_in[8];
  const float* g_ln = (const float*)d_in[9];
  const float* be_ln= (const float*)d_in[10];
  const float* Wg   = (const float*)d_in[11];
  const float* bg   = (const float*)d_in[12];
  const float* Wt1  = (const float*)d_in[13];
  const float* bt1  = (const float*)d_in[14];
  const float* Wt2  = (const float*)d_in[15];
  const float* bt2  = (const float*)d_in[16];
  float* outp = (float*)d_out;

  char* ws = (char*)d_ws;
  size_t off = 0;
  auto take = [&](size_t bytes) {
    void* p = ws + off;
    off += (bytes + 255) & ~(size_t)255;
    return p;
  };
  _Float16* h16     = (_Float16*)take((size_t)Bsz * D * 2);
  _Float16* out_all = (_Float16*)take((size_t)Bsz * E * D * 2);
  float*    wgate   = (float*)take((size_t)Bsz * T * G * 4);
  _Float16* x16     = (_Float16*)take((size_t)Bsz * F * 2);
  _Float16* Wpt     = (_Float16*)take((size_t)F * D * 2);
  _Float16* W1t     = (_Float16*)take((size_t)LVL * E * Hd * D * 2);
  _Float16* W2t     = (_Float16*)take((size_t)LVL * E * D * Hd * 2);
  _Float16* Wt1t    = (_Float16*)take((size_t)T * THd * D * 2);
  _Float16* Wt2t    = (_Float16*)take((size_t)T * OP * THd * 2);

  dim3 block(256);
  // conversions
  {
    int n = Bsz * F;
    k_cvt_copy<<<(n + 255) / 256, block, 0, stream>>>(x, x16, n);
    n = F * D;
    k_cvt_tr<<<(n + 255) / 256, block, 0, stream>>>(Wp, Wpt, 1, F, D);
    n = LVL * E * D * Hd;
    k_cvt_tr<<<(n + 255) / 256, block, 0, stream>>>(W1, W1t, LVL * E, D, Hd);
    k_cvt_tr<<<(n + 255) / 256, block, 0, stream>>>(W2, W2t, LVL * E, Hd, D);
    n = T * D * THd;
    k_cvt_tr<<<(n + 255) / 256, block, 0, stream>>>(Wt1, Wt1t, T, D, THd);
    n = T * OP * THd;
    k_cvt_wt2<<<(n + 255) / 256, block, 0, stream>>>(Wt2, Wt2t);
  }

  dim3 grid(Bsz / MB);
  k_proj<<<grid, block, 0, stream>>>(x16, Wpt, bp, gp, bep, h16);

  for (int l = 0; l < LVL; ++l) {
    k_ple<0><<<grid, block, 0, stream>>>(
        W1t + (size_t)l * E * Hd * D, b1 + (size_t)l * E * Hd,
        W2t + (size_t)l * E * D * Hd, b2 + (size_t)l * E * D,
        g_ln + (size_t)l * E * D, be_ln + (size_t)l * E * D,
        Wg + (size_t)l * T * D * G, bg + (size_t)l * T * G,
        h16, out_all, wgate);
  }
  const int lf = LVL - 1;
  k_ple<1><<<grid, block, 0, stream>>>(
      W1t + (size_t)lf * E * Hd * D, b1 + (size_t)lf * E * Hd,
      W2t + (size_t)lf * E * D * Hd, b2 + (size_t)lf * E * D,
      g_ln + (size_t)lf * E * D, be_ln + (size_t)lf * E * D,
      Wg + (size_t)lf * T * D * G, bg + (size_t)lf * T * G,
      h16, out_all, wgate);

  k_heads<<<grid, block, 0, stream>>>(out_all, wgate, Wt1t, bt1, Wt2t, bt2, outp);
}
